// DecoderLayer_77841987273024
// MI455X (gfx1250) — matrix operand order repaired and checked by reading, not compiled, not run on hardware
//
#include <hip/hip_runtime.h>
#include <math.h>

// ---------------- problem constants (match reference) ----------------
constexpr int Ltok = 2048;
constexpr int Dm   = 2048;
constexpr int HK   = 8,  HV = 16, DK = 128, DV = 128;
constexpr int KCONV = 4;
constexpr int Emoe = 8, TOPK = 2, IMOE = 768, ISH = 4096;
constexpr int KEY_DIM  = HK * DK;            // 1024
constexpr int VAL_DIM  = HV * DV;            // 2048
constexpr int CONV_DIM = 2 * KEY_DIM + VAL_DIM; // 4096
constexpr float EPS = 1e-6f;

typedef __bf16 bf16;
typedef __attribute__((ext_vector_type(16))) __bf16 v16bf;
typedef __attribute__((ext_vector_type(8)))  __bf16 v8bf;
typedef __attribute__((ext_vector_type(4)))  __bf16 v4bf;
typedef __attribute__((ext_vector_type(8)))  float  v8f;

typedef __attribute__((ext_vector_type(4))) unsigned int u32x4;
typedef __attribute__((ext_vector_type(8))) int          i32x8;
typedef __attribute__((ext_vector_type(4))) int          i32x4;

union V16U { v16bf v; v8bf h[2]; };

#if defined(__has_builtin)
#if __has_builtin(__builtin_amdgcn_tensor_load_to_lds)
#define HAVE_TDM 1
#endif
#endif
#ifndef HAVE_TDM
#define HAVE_TDM 0
#endif

#if HAVE_TDM
__device__ __forceinline__ void tdm_issue_2d(u32x4 g0, i32x8 g1)
{
    i32x4 z4 = {0, 0, 0, 0};
#if __clang_major__ >= 23
    i32x8 z8 = {0, 0, 0, 0, 0, 0, 0, 0};
    __builtin_amdgcn_tensor_load_to_lds(g0, g1, z4, z4, z8, 0);
#else
    __builtin_amdgcn_tensor_load_to_lds(g0, g1, z4, z4, 0);
#endif
}
#endif

__device__ __forceinline__ float siluf(float x) { return x / (1.f + __expf(-x)); }
__device__ __forceinline__ float sigmf(float x) { return 1.f / (1.f + __expf(-x)); }

// =====================================================================
// Generic bf16 WMMA GEMM:  C[N,M] (f32) = A[N,K] (bf16) * W[M,K]^T (f32->bf16)
// Block tile 128x128, 8 waves, wave tile 32(tokens) x 64(features).
// A tile staged by the Tensor Data Mover (incl. LDS row padding); W tile
// staged manually with inline fp32->bf16 conversion.
// =====================================================================
__global__ __launch_bounds__(256)
void k_gemm_bf16(const bf16* __restrict__ A, const float* __restrict__ W,
                 float* __restrict__ C, int Nt, int M, int K, int addC)
{
    __shared__ __align__(16) bf16 As[128][40];   // 80B row stride (TDM pad: 64B + 16B)
    __shared__ __align__(16) bf16 Ws[128][40];

    const int tid  = threadIdx.x;
    const int lane = tid & 31, wid = tid >> 5;
    const int grp  = lane >> 4, l16 = lane & 15;
    const int tokBase  = blockIdx.y * 128;
    const int featBase = blockIdx.x * 128;
    const int waveTok  = (wid & 3) * 32;      // 4 waves along tokens
    const int waveFeat = (wid >> 2) * 64;     // 2 waves along features

    v8f acc[2][4];
#pragma unroll
    for (int i = 0; i < 2; ++i)
#pragma unroll
        for (int j = 0; j < 4; ++j)
#pragma unroll
            for (int r = 0; r < 8; ++r) acc[i][j][r] = 0.f;

#if HAVE_TDM
    // ---- TDM descriptor group 1 (loop-invariant): bf16 elements (2B),
    // tensor = [Nt rows x K cols], tile = 128 x 32, row stride = K,
    // pad_enable, pad after every 16 DWORDs (one 64B tile row), pad 4 DWORDs.
    const unsigned td0 = (unsigned)K, td1 = (unsigned)Nt;
    const unsigned long long str0 = (unsigned long long)K;
    i32x8 g1;
    g1[0] = (int)((1u << 16) | (1u << 20) | (3u << 22) | (3u << 25));
    g1[1] = (int)((td0 & 0xFFFFu) << 16);
    g1[2] = (int)((td0 >> 16) | ((td1 & 0xFFFFu) << 16));
    g1[3] = (int)((td1 >> 16) | (32u << 16));                 // tile_dim0 = 32
    g1[4] = (int)(128u);                                       // tile_dim1 = 128, tile_dim2 = 0
    g1[5] = (int)(unsigned)(str0 & 0xFFFFFFFFu);
    g1[6] = (int)(unsigned)((str0 >> 32) & 0xFFFFu);           // stride1 = 0
    g1[7] = 0;
    const unsigned lds_a = (unsigned)(size_t)(&As[0][0]);
    const unsigned long long gbaseA =
        (unsigned long long)(size_t)A + (unsigned long long)tokBase * (unsigned)K * 2ull;
#endif

    for (int k0 = 0; k0 < K; k0 += 32) {
        // ---- stage A tile (bf16, 128x32) ----
#if HAVE_TDM
        if (wid == 0) {
            unsigned long long ga = gbaseA + (unsigned long long)k0 * 2ull;
            u32x4 g0;
            g0[0] = 1u;                                        // count=1, user mode
            g0[1] = lds_a;                                     // LDS byte address
            g0[2] = (unsigned)(ga & 0xFFFFFFFFu);              // global_addr[31:0]
            g0[3] = (unsigned)((ga >> 32) & 0x01FFFFFFu) | (2u << 30);  // addr hi | type=2
            tdm_issue_2d(g0, g1);
            __builtin_amdgcn_s_wait_tensorcnt((short)0);
        }
#else
#pragma unroll
        for (int i = tid; i < 512; i += 256) {
            int r = i >> 2, cc = (i & 3) << 3;
            int gr = tokBase + r;
            v8bf val;
            if (gr < Nt) {
                val = *(const v8bf*)(A + (size_t)gr * K + k0 + cc);
            } else {
#pragma unroll
                for (int j = 0; j < 8; ++j) val[j] = (bf16)0.0f;
            }
            *(v8bf*)&As[r][cc] = val;
        }
#endif
        // ---- stage W tile (fp32 -> bf16, 128x32) ----
#pragma unroll
        for (int i = tid; i < 1024; i += 256) {
            int r = i >> 3, cc = (i & 7) << 2;
            int gr = featBase + r;
            float4 f = make_float4(0.f, 0.f, 0.f, 0.f);
            if (gr < M) {
                f = *(const float4*)(W + (size_t)gr * K + k0 + cc);
                if (k0 + 64 < K) __builtin_prefetch(W + (size_t)gr * K + k0 + 64, 0, 1);
            }
            v4bf o;
            o[0] = (bf16)f.x; o[1] = (bf16)f.y; o[2] = (bf16)f.z; o[3] = (bf16)f.w;
            *(v4bf*)&Ws[r][cc] = o;
        }
        __syncthreads();

        // ---- fragments per documented CDNA5 layouts ----
        V16U af[2], bfr[4];
#pragma unroll
        for (int s = 0; s < 2; ++s) {
            int r = waveTok + s * 16 + l16;
            af[s].h[0] = *(const v8bf*)&As[r][grp * 8];         // halves 0-7 : K=8g+j
            af[s].h[1] = *(const v8bf*)&As[r][16 + grp * 8];    // halves 8-15: K=16+8g+(j-8)
        }
#pragma unroll
        for (int s = 0; s < 4; ++s) {
            int r = waveFeat + s * 16 + l16;                    // B col = feature row of W
            bfr[s].h[0] = *(const v8bf*)&Ws[r][grp * 8];        // halves 0-7 : K=8g+j
            bfr[s].h[1] = *(const v8bf*)&Ws[r][16 + grp * 8];   // halves 8-15: K=16+8g+(j-8)
        }
#pragma unroll
        for (int si = 0; si < 2; ++si)
#pragma unroll
            for (int sj = 0; sj < 4; ++sj)
                acc[si][sj] = __builtin_amdgcn_wmma_f32_16x16x32_bf16(
                    false, af[si].v, false, bfr[sj].v,
                    (short)0, acc[si][sj], false, false);
        __syncthreads();
    }

    // ---- epilogue: C/D layout lane=(N), vgpr r -> M = 8*grp + r ----
#pragma unroll
    for (int sj = 0; sj < 4; ++sj) {
        int col = featBase + waveFeat + sj * 16 + l16;
        if (col >= M) continue;
#pragma unroll
        for (int si = 0; si < 2; ++si) {
#pragma unroll
            for (int r = 0; r < 8; ++r) {
                int row = tokBase + waveTok + si * 16 + grp * 8 + r;
                if (row < Nt) {
                    size_t idx = (size_t)row * M + col;
                    float v = acc[si][sj][r];
                    C[idx] = addC ? (C[idx] + v) : v;
                }
            }
        }
    }
}

// =====================================================================
// OffsetRMSNorm: h = (1+w) * x * rsqrt(mean(x^2)+eps)  -> bf16
// =====================================================================
__global__ __launch_bounds__(256)
void k_rms(const float* __restrict__ x, const float* __restrict__ w, bf16* __restrict__ out)
{
    __shared__ float red[256];
    int t = blockIdx.x, tid = threadIdx.x;
    float vals[8]; float ss = 0.f;
#pragma unroll
    for (int i = 0; i < 8; ++i) {
        vals[i] = x[(size_t)t * Dm + tid + i * 256];
        ss += vals[i] * vals[i];
    }
    red[tid] = ss; __syncthreads();
    for (int s = 128; s > 0; s >>= 1) { if (tid < s) red[tid] += red[tid + s]; __syncthreads(); }
    float inv = rsqrtf(red[0] * (1.f / Dm) + EPS);
#pragma unroll
    for (int i = 0; i < 8; ++i) {
        int c = tid + i * 256;
        out[(size_t)t * Dm + c] = (bf16)((1.f + w[c]) * vals[i] * inv);
    }
}

// residual + OffsetRMSNorm: x2 = x + y1 ; h2 = rms(x2, w)
__global__ __launch_bounds__(256)
void k_resid_rms(const float* __restrict__ x, const float* __restrict__ y1,
                 const float* __restrict__ w, float* __restrict__ x2, bf16* __restrict__ h2)
{
    __shared__ float red[256];
    int t = blockIdx.x, tid = threadIdx.x;
    float vals[8]; float ss = 0.f;
#pragma unroll
    for (int i = 0; i < 8; ++i) {
        size_t idx = (size_t)t * Dm + tid + i * 256;
        float v = x[idx] + y1[idx];
        x2[idx] = v; vals[i] = v; ss += v * v;
    }
    red[tid] = ss; __syncthreads();
    for (int s = 128; s > 0; s >>= 1) { if (tid < s) red[tid] += red[tid + s]; __syncthreads(); }
    float inv = rsqrtf(red[0] * (1.f / Dm) + EPS);
#pragma unroll
    for (int i = 0; i < 8; ++i) {
        int c = tid + i * 256;
        h2[(size_t)t * Dm + c] = (bf16)((1.f + w[c]) * vals[i] * inv);
    }
}

// =====================================================================
// depthwise causal conv1d (K=4) + SiLU, per-sequence reset
// =====================================================================
__global__ __launch_bounds__(256)
void k_conv_silu(const float* __restrict__ qkv, const float* __restrict__ cw,
                 const int* __restrict__ cu, int ncu, float* __restrict__ mixed)
{
    int idx = blockIdx.x * 256 + threadIdx.x;
    if (idx >= Ltok * CONV_DIM) return;
    int t = idx / CONV_DIM, c = idx - t * CONV_DIM;
    int start = 0;
    for (int j = 0; j < ncu - 1; ++j) { int s = cu[j]; if (s <= t) start = s; }
    float y = 0.f;
#pragma unroll
    for (int j = 0; j < KCONV; ++j) {
        int ts = t - (KCONV - 1 - j);
        if (ts >= start) y += qkv[(size_t)ts * CONV_DIM + c] * cw[c * KCONV + j];
    }
    mixed[idx] = siluf(y);
}

// q/k L2 norm per head; q gets DK^-0.5 folded in. blockIdx=(t, s) s<8->q head, else k head.
__global__ __launch_bounds__(128)
void k_qknorm(const float* __restrict__ mixed, float* __restrict__ qn, float* __restrict__ kn)
{
    __shared__ float red[128];
    int t = blockIdx.x, s = blockIdx.y, tid = threadIdx.x;
    bool isQ = s < HK; int h = isQ ? s : s - HK;
    const float* src = mixed + (size_t)t * CONV_DIM + (isQ ? 0 : KEY_DIM) + h * DK;
    float v = src[tid];
    red[tid] = v * v; __syncthreads();
    for (int st = 64; st > 0; st >>= 1) { if (tid < st) red[tid] += red[tid + st]; __syncthreads(); }
    float o = v * rsqrtf(red[0] + EPS);
    if (isQ) o *= 0.08838834764831845f;    // DK^-0.5
    (isQ ? qn : kn)[(size_t)t * KEY_DIM + h * DK + tid] = o;
}

// g = -exp(A_log)*softplus(a+dt_bias), beta = sigmoid(b)
__global__ __launch_bounds__(256)
void k_gab(const float* __restrict__ a, const float* __restrict__ b,
           const float* __restrict__ Alog, const float* __restrict__ dtb,
           float* __restrict__ g, float* __restrict__ be)
{
    int idx = blockIdx.x * 256 + threadIdx.x;
    if (idx >= Ltok * HV) return;
    int h = idx & (HV - 1);
    float x = a[idx] + dtb[h];
    float sp = (x > 20.f) ? x : log1pf(__expf(x));
    g[idx]  = -__expf(Alog[h]) * sp;
    be[idx] = sigmf(b[idx]);
}

// =====================================================================
// delta-rule scan: one block per head, one thread per dv column,
// per-thread 128-deep dk state held in VGPRs.
// =====================================================================
__global__ __launch_bounds__(128)
void k_scan(const float* __restrict__ qn, const float* __restrict__ kn,
            const float* __restrict__ mixed, const float* __restrict__ g,
            const float* __restrict__ be, const int* __restrict__ cu, int ncu,
            float* __restrict__ o)
{
    int h = blockIdx.x, tid = threadIdx.x;  // tid = dv
    int kh = h >> 1;                        // rep = HV/HK = 2
    __shared__ float lk[DK], lq[DK];
    __shared__ int scu[64];
    if (tid < ncu && tid < 64) scu[tid] = cu[tid];
    __syncthreads();

    float S[DK];
#pragma unroll
    for (int i = 0; i < DK; ++i) S[i] = 0.f;

    for (int t = 0; t < Ltok; ++t) {
        lk[tid] = kn[(size_t)t * KEY_DIM + kh * DK + tid];
        lq[tid] = qn[(size_t)t * KEY_DIM + kh * DK + tid];
        __syncthreads();
        bool rst = false;
        for (int j = 0; j < ncu - 1; ++j) rst |= (scu[j] == t);
        float gt = g[t * HV + h], bt = be[t * HV + h];
        float dec = rst ? 0.f : __expf(gt);
        float kv = 0.f;
#pragma unroll
        for (int i = 0; i < DK; ++i) { S[i] *= dec; kv += lk[i] * S[i]; }
        float vt = mixed[(size_t)t * CONV_DIM + 2 * KEY_DIM + h * DV + tid];
        float delta = bt * (vt - kv);
        float oacc = 0.f;
#pragma unroll
        for (int i = 0; i < DK; ++i) { S[i] += lk[i] * delta; oacc += lq[i] * S[i]; }
        o[(size_t)t * VAL_DIM + h * DV + tid] = oacc;
        __syncthreads();
    }
}

// gated RMSNorm with swish gate -> bf16 activation for out-proj
__global__ __launch_bounds__(128)
void k_gnorm(const float* __restrict__ o, const float* __restrict__ z,
             const float* __restrict__ gw, bf16* __restrict__ attn)
{
    __shared__ float red[128];
    int t = blockIdx.x, h = blockIdx.y, tid = threadIdx.x;
    size_t idx = (size_t)t * VAL_DIM + h * DV + tid;
    float ov = o[idx];
    red[tid] = ov * ov; __syncthreads();
    for (int s = 64; s > 0; s >>= 1) { if (tid < s) red[tid] += red[tid + s]; __syncthreads(); }
    float inv = rsqrtf(red[0] * (1.f / DV) + EPS);
    attn[idx] = (bf16)(ov * inv * gw[tid] * siluf(z[idx]));
}

// =====================================================================
// MoE helpers
// =====================================================================
__global__ void k_init_aux(float* auxb) { if (threadIdx.x < 16) auxb[threadIdx.x] = 0.f; }

__global__ __launch_bounds__(256)
void k_router(const float* __restrict__ logits, float* __restrict__ comb, float* __restrict__ auxb)
{
    int n = blockIdx.x * 256 + threadIdx.x;
    if (n >= Ltok) return;
    float p[Emoe]; float m = -1e30f;
#pragma unroll
    for (int e = 0; e < Emoe; ++e) { p[e] = logits[n * Emoe + e]; m = fmaxf(m, p[e]); }
    float s = 0.f;
#pragma unroll
    for (int e = 0; e < Emoe; ++e) { p[e] = __expf(p[e] - m); s += p[e]; }
    float psum = 0.f;
#pragma unroll
    for (int e = 0; e < Emoe; ++e) { p[e] /= s; psum += p[e]; }
    int i1 = 0;
#pragma unroll
    for (int e = 1; e < Emoe; ++e) if (p[e] > p[i1]) i1 = e;
    int i2 = (i1 == 0) ? 1 : 0;
#pragma unroll
    for (int e = 0; e < Emoe; ++e) if (e != i1 && p[e] > p[i2]) i2 = e;
    float tot = p[i1] + p[i2];
#pragma unroll
    for (int e = 0; e < Emoe; ++e) comb[n * Emoe + e] = 0.f;
    comb[n * Emoe + i1] = p[i1] / tot;
    comb[n * Emoe + i2] = p[i2] / tot;
    atomicAdd(&auxb[i1], 1.f);
    atomicAdd(&auxb[i2], 1.f);
    atomicAdd(&auxb[Emoe], psum);
}

__global__ __launch_bounds__(256)
void k_glu_sh(const float* __restrict__ gg, const float* __restrict__ uu, bf16* __restrict__ he, int n)
{
    int i = blockIdx.x * 256 + threadIdx.x;
    if (i < n) he[i] = (bf16)(siluf(gg[i]) * uu[i]);
}

__global__ __launch_bounds__(256)
void k_glu_moe(const float* __restrict__ gu, const float* __restrict__ comb, int e,
               bf16* __restrict__ he)
{
    int i = blockIdx.x * 256 + threadIdx.x;
    if (i >= Ltok * IMOE) return;
    int n = i / IMOE, c = i - n * IMOE;
    float gg = gu[(size_t)n * (2 * IMOE) + c];
    float uu = gu[(size_t)n * (2 * IMOE) + IMOE + c];
    he[i] = (bf16)(siluf(gg) * uu * comb[n * Emoe + e]);
}

__global__ __launch_bounds__(256)
void k_final(const float* __restrict__ x2, const float* __restrict__ y,
             const float* __restrict__ sh, const float* __restrict__ eg,
             float* __restrict__ out)
{
    int i = blockIdx.x * 256 + threadIdx.x;
    if (i >= Ltok * Dm) return;
    int n = i >> 11;
    out[i] = x2[i] + y[i] + sigmf(eg[n]) * sh[i];
}

__global__ void k_aux(const float* __restrict__ auxb, float* __restrict__ out_aux)
{
    if (threadIdx.x != 0 || blockIdx.x != 0) return;
    float frac_p = auxb[Emoe] / (float)Ltok;
    float s = 0.f;
    for (int e = 0; e < Emoe; ++e) s += (auxb[e] / (float)(Ltok * TOPK)) * frac_p;
    *out_aux = (float)Emoe * s;
}

// =====================================================================
// host-side orchestration
// =====================================================================
extern "C" void kernel_launch(void* const* d_in, const int* in_sizes, int n_in,
                              void* d_out, int out_size, void* d_ws, size_t ws_size,
                              hipStream_t stream)
{
    (void)n_in; (void)out_size; (void)ws_size;
    const float* x        = (const float*)d_in[0];
    const int*   cu       = (const int*)  d_in[1];
    const float* w_qkv    = (const float*)d_in[2];
    const float* w_z      = (const float*)d_in[3];
    const float* w_a      = (const float*)d_in[4];
    const float* w_b      = (const float*)d_in[5];
    const float* conv_w   = (const float*)d_in[6];
    const float* A_log    = (const float*)d_in[7];
    const float* dt_bias  = (const float*)d_in[8];
    const float* gnorm_w  = (const float*)d_in[9];
    const float* w_out    = (const float*)d_in[10];
    const float* ln1_w    = (const float*)d_in[11];
    const float* ln2_w    = (const float*)d_in[12];
    const float* router_w = (const float*)d_in[13];
    const float* gate_up  = (const float*)d_in[14];
    const float* down_p   = (const float*)d_in[15];
    const float* sh_gate  = (const float*)d_in[16];
    const float* sh_up    = (const float*)d_in[17];
    const float* sh_down  = (const float*)d_in[18];
    const float* sh_eg    = (const float*)d_in[19];
    float* out = (float*)d_out;

    const int ncu = in_sizes[1];
    char* wsb = (char*)d_ws;
    const size_t MB = 1ull << 20;
    auto F  = [&](size_t off) { return (float*)(wsb + off); };
    auto BF = [&](size_t off) { return (bf16*)(wsb + off); };

    // persistent
    float* x2    = F(0);                 // 16MB
    bf16*  h2    = BF(16 * MB);          // 8MB
    float* logits= F(24 * MB);
    float* comb  = F(24 * MB + 256 * 1024);
    float* eg    = F(24 * MB + 512 * 1024);
    float* auxb  = F(24 * MB + 768 * 1024);
    const size_t S0 = 25 * MB;
    // phase 1 (GDN)
    bf16*  hbf   = BF(S0);
    float* qkv   = F(S0 + 8 * MB);
    float* z     = F(S0 + 40 * MB);
    float* a     = F(S0 + 56 * MB);
    float* b     = F(S0 + 57 * MB);
    float* g     = F(S0 + 58 * MB);
    float* be    = F(S0 + 59 * MB);
    float* mixed = F(S0 + 60 * MB);
    float* qn    = F(S0 + 92 * MB);
    float* kn    = F(S0 + 100 * MB);
    float* o     = F(S0 + 108 * MB);
    bf16*  attn  = BF(S0 + 124 * MB);
    float* y1    = F(S0 + 132 * MB);
    // phase 2 (MoE) — reuses phase-1 scratch (all dead by then)
    float* shg   = F(S0);
    float* shu   = F(S0 + 32 * MB);
    bf16*  hesh  = BF(S0 + 64 * MB);
    float* sh    = F(S0 + 80 * MB);
    float* gu    = F(S0 + 96 * MB);
    bf16*  he    = BF(S0 + 108 * MB);
    float* y     = F(S0 + 112 * MB);

    auto gemm = [&](const bf16* A, const float* W, float* C, int Nt, int M, int K, int add) {
        dim3 gr((M + 127) / 128, (Nt + 127) / 128);
        k_gemm_bf16<<<gr, 256, 0, stream>>>(A, W, C, Nt, M, K, add);
    };

    // ---------------- GDN ----------------
    k_rms<<<Ltok, 256, 0, stream>>>(x, ln1_w, hbf);
    gemm(hbf, w_qkv, qkv, Ltok, CONV_DIM, Dm, 0);
    gemm(hbf, w_z,   z,   Ltok, VAL_DIM,  Dm, 0);
    gemm(hbf, w_a,   a,   Ltok, HV,       Dm, 0);
    gemm(hbf, w_b,   b,   Ltok, HV,       Dm, 0);
    k_conv_silu<<<(Ltok * CONV_DIM) / 256, 256, 0, stream>>>(qkv, conv_w, cu, ncu, mixed);
    k_qknorm<<<dim3(Ltok, 2 * HK), 128, 0, stream>>>(mixed, qn, kn);
    k_gab<<<(Ltok * HV) / 256, 256, 0, stream>>>(a, b, A_log, dt_bias, g, be);
    k_scan<<<HV, 128, 0, stream>>>(qn, kn, mixed, g, be, cu, ncu, o);
    k_gnorm<<<dim3(Ltok, HV), 128, 0, stream>>>(o, z, gnorm_w, attn);
    gemm(attn, w_out, y1, Ltok, Dm, VAL_DIM, 0);
    k_resid_rms<<<Ltok, 256, 0, stream>>>(x, y1, ln2_w, x2, h2);

    // ---------------- MoE ----------------
    gemm(h2, router_w, logits, Ltok, Emoe, Dm, 0);
    k_init_aux<<<1, 32, 0, stream>>>(auxb);
    k_router<<<Ltok / 256, 256, 0, stream>>>(logits, comb, auxb);

    // shared expert
    gemm(h2, sh_gate, shg, Ltok, ISH, Dm, 0);
    gemm(h2, sh_up,   shu, Ltok, ISH, Dm, 0);
    k_glu_sh<<<(Ltok * ISH) / 256, 256, 0, stream>>>(shg, shu, hesh, Ltok * ISH);
    gemm(hesh, sh_down, sh, Ltok, Dm, ISH, 0);
    gemm(h2, sh_eg, eg, Ltok, 1, Dm, 0);

    // routed experts (dense-masked, reference math; scale folded into activations)
    for (int e = 0; e < Emoe; ++e) {
        gemm(h2, gate_up + (size_t)e * 2 * IMOE * Dm, gu, Ltok, 2 * IMOE, Dm, 0);
        k_glu_moe<<<(Ltok * IMOE) / 256, 256, 0, stream>>>(gu, comb, e, he);
        gemm(he, down_p + (size_t)e * Dm * IMOE, y, Ltok, Dm, IMOE, e > 0 ? 1 : 0);
    }

    k_final<<<(Ltok * Dm) / 256, 256, 0, stream>>>(x2, y, sh, eg, out);
    k_aux<<<1, 32, 0, stream>>>(auxb, out + (size_t)Ltok * Dm);
}
